// EIIERecurrent_76192719832117
// MI455X (gfx1250) — matrix operand order repaired and checked by reading, not compiled, not run on hardware
//
#include <hip/hip_runtime.h>
#include <hip/hip_bf16.h>
#include <math.h>

typedef __attribute__((ext_vector_type(16))) _Float16 v16h;
typedef __attribute__((ext_vector_type(2)))  _Float16 v2h;
typedef __attribute__((ext_vector_type(2)))  __fp16   v2fp16;
typedef __attribute__((ext_vector_type(8)))  float    v8f;

#define PP 11
#define BB 64
#define TT 512
#define FF 3
#define HH 128
#define LL 20
#define BSPLIT 2
#define BW 32            // batch rows per workgroup

// ---- LDS layout (bytes) ----
#define OFF_H     0                          // h_state: LL*BW*HH f16  = 163840
#define OFF_X0    (OFF_H    + LL*BW*HH*2)    // layer-0 input proj f32 = 16384
#define OFF_WIH   (OFF_X0   + BW*HH*4)       // W_ih f16 (128x128)     = 32768
#define OFF_WHH   (OFF_WIH  + HH*HH*2)       // W_hh f16 (128x128)     = 32768
#define OFF_BIAS  (OFF_WHH  + HH*HH*2)       // biases f32 LL*HH       = 10240
#define OFF_WIH0  (OFF_BIAS + LL*HH*4)       // w_ih0 f32 HH*FF        = 1536
#define LDS_BYTES (OFF_WIH0 + HH*FF*4)       // = 257536  (<= 320KB WGP LDS)

union AFrag { v16h v; uint4 q[2]; };
union H4    { _Float16 h[4]; v2h p[2]; uint2 u; };

// Native CDNA5 v_tanh_f32 (single TRANS32 op, co-executes with WMMA);
// branch-free exp2/rcp fallback if the builtin is unavailable.
static __device__ __forceinline__ float fast_tanh(float x)
{
#if __has_builtin(__builtin_amdgcn_tanhf)
    return __builtin_amdgcn_tanhf(x);
#elif __has_builtin(__builtin_amdgcn_tanh_f32)
    return __builtin_amdgcn_tanh_f32(x);
#else
    float xc = fminf(fmaxf(x, -10.0f), 10.0f);
    float e  = __builtin_amdgcn_exp2f(2.8853900817779268f * xc);  // exp(2x)
    return (e - 1.0f) * __builtin_amdgcn_rcpf(e + 1.0f);
#endif
}

// Pack two f32 -> two f16 with one v_cvt_pk_rtz_f16_f32 when available.
static __device__ __forceinline__ v2h pack_f16(float a, float b)
{
#if __has_builtin(__builtin_amdgcn_cvt_pkrtz)
    union { v2fp16 raw; v2h out; } cvt;
    cvt.raw = __builtin_amdgcn_cvt_pkrtz(a, b);
    return cvt.out;
#else
    v2h r; r[0] = (_Float16)a; r[1] = (_Float16)b; return r;
#endif
}

__global__ __launch_bounds__(256, 1)
void eiie_rnn_kernel(const float* __restrict__ obs,        // [B,F,P,T]
                     const float* __restrict__ last_action,// [B,P+1]
                     const float* __restrict__ w_ih0,      // [P,H,F]
                     const float* __restrict__ w_ih,       // [P,L-1,H,H]
                     const float* __restrict__ w_hh,       // [P,L,H,H]
                     const float* __restrict__ b_ih,       // [P,L,H]
                     const float* __restrict__ b_hh,       // [P,L,H]
                     const float* __restrict__ conv_w,     // [1,H+1]
                     const float* __restrict__ conv_b,     // [1]
                     float* __restrict__ out)              // [B,P+1] (logits slots 1..P)
{
    extern __shared__ char smem[];
    _Float16* hS    = (_Float16*)(smem + OFF_H);
    float*    x0    = (float*)   (smem + OFF_X0);
    _Float16* wihL  = (_Float16*)(smem + OFF_WIH);
    _Float16* whhL  = (_Float16*)(smem + OFF_WHH);
    float*    biasL = (float*)   (smem + OFF_BIAS);
    float*    wih0L = (float*)   (smem + OFF_WIH0);

    const int tid  = threadIdx.x;
    const int p    = blockIdx.x / BSPLIT;   // asset
    const int bs   = blockIdx.x % BSPLIT;   // batch half
    const int wave = tid >> 5;              // 0..7 -> owns N-tile = wave
    const int lane = tid & 31;
    const int lo   = lane & 15;
    const int hi   = lane >> 4;

    const float* whh_g = w_hh + (size_t)p * LL * HH * HH;
    const float* wih_g = w_ih + (size_t)p * (LL - 1) * HH * HH;

    // ---- init: zero hidden states, stage biases (b_ih+b_hh) and w_ih0 ----
    {
        uint4 z; z.x = z.y = z.z = z.w = 0u;
        uint4* hz = (uint4*)hS;
        for (int i = tid; i < (LL * BW * HH * 2) / 16; i += 256) hz[i] = z;
        for (int i = tid; i < LL * HH; i += 256)
            biasL[i] = b_ih[(size_t)p * LL * HH + i] + b_hh[(size_t)p * LL * HH + i];
        for (int i = tid; i < HH * FF; i += 256)
            wih0L[i] = w_ih0[(size_t)p * HH * FF + i];
    }
    __syncthreads();

    const int n = wave * 16 + lo;           // this lane's output column (both tiles)

    for (int t = 0; t < TT; ++t) {
        for (int l = 0; l < LL; ++l) {
            __syncthreads();   // prior layer stores done; weight buffers free

            // ---- stage layer weights fp32 -> f16 into LDS ----
            {
                const float4* src = (const float4*)(whh_g + (size_t)l * HH * HH);
                #pragma unroll
                for (int j = 0; j < 16; ++j) {
                    int i = tid + j * 256;            // float4 index (4096 total)
                    float4 v = src[i];
                    H4 c;
                    c.p[0] = pack_f16(v.x, v.y);
                    c.p[1] = pack_f16(v.z, v.w);
                    *(uint2*)(whhL + (size_t)i * 4) = c.u;
                }
                if (l > 0) {
                    const float4* s2 = (const float4*)(wih_g + (size_t)(l - 1) * HH * HH);
                    #pragma unroll
                    for (int j = 0; j < 16; ++j) {
                        int i = tid + j * 256;
                        float4 v = s2[i];
                        H4 c;
                        c.p[0] = pack_f16(v.x, v.y);
                        c.p[1] = pack_f16(v.z, v.w);
                        *(uint2*)(wihL + (size_t)i * 4) = c.u;
                    }
                } else {
                    // layer-0 input projection (F=3, VALU) incl. bias
                    #pragma unroll
                    for (int j = 0; j < 16; ++j) {
                        int e = tid + j * 256;        // 4096 = 32*128
                        int m = e >> 7, nn = e & 127;
                        int bg = bs * BW + m;
                        float xi = biasL[nn];
                        #pragma unroll
                        for (int f = 0; f < FF; ++f)
                            xi += obs[((size_t)bg * FF + f) * (PP * TT) + (size_t)p * TT + t]
                                  * wih0L[nn * FF + f];
                        x0[e] = xi;
                    }
                }
            }
            __syncthreads();   // weights / x0 visible

            // ---- GEMMs: acc = bias + x_in@Wih^T + h_prev@Whh^T ----
            _Float16* hcur = hS + (size_t)l * BW * HH;
            const _Float16* hin = (l > 0) ? (hS + (size_t)(l - 1) * BW * HH) : (const _Float16*)0;

            v8f acc0 = {}, acc1 = {};
            if (l == 0) {
                #pragma unroll
                for (int r = 0; r < 8; ++r) {
                    acc0[r] = x0[(hi * 8 + r) * HH + n];
                    acc1[r] = x0[(16 + hi * 8 + r) * HH + n];
                }
            } else {
                float bv = biasL[l * HH + n];
                #pragma unroll
                for (int r = 0; r < 8; ++r) { acc0[r] = bv; acc1[r] = bv; }
            }

            #pragma unroll
            for (int k0 = 0; k0 < 4; ++k0) {
                AFrag bhh, a0, a1;
                // B = Whh^T: lane holds two 8-half K chunks of row n (row-major [n][k])
                const uint4* bp = (const uint4*)(whhL + (size_t)n * HH + k0 * 32 + hi * 8);
                bhh.q[0] = bp[0]; bhh.q[1] = bp[2];
                // A = h_prev (old state of this layer), rows m, two 8-half chunks
                const _Float16* ap0 = hcur + (size_t)lo * HH + k0 * 32 + hi * 8;
                a0.q[0] = *(const uint4*)ap0;
                a0.q[1] = *(const uint4*)(ap0 + 16);
                const _Float16* ap1 = hcur + (size_t)(16 + lo) * HH + k0 * 32 + hi * 8;
                a1.q[0] = *(const uint4*)ap1;
                a1.q[1] = *(const uint4*)(ap1 + 16);

                acc0 = __builtin_amdgcn_wmma_f32_16x16x32_f16(false, a0.v, false, bhh.v,
                                                              (short)0, acc0, false, false);
                acc1 = __builtin_amdgcn_wmma_f32_16x16x32_f16(false, a1.v, false, bhh.v,
                                                              (short)0, acc1, false, false);
                if (l > 0) {
                    AFrag bih, ai0, ai1;
                    const uint4* bq = (const uint4*)(wihL + (size_t)n * HH + k0 * 32 + hi * 8);
                    bih.q[0] = bq[0]; bih.q[1] = bq[2];
                    const _Float16* aq0 = hin + (size_t)lo * HH + k0 * 32 + hi * 8;
                    ai0.q[0] = *(const uint4*)aq0;
                    ai0.q[1] = *(const uint4*)(aq0 + 16);
                    const _Float16* aq1 = hin + (size_t)(16 + lo) * HH + k0 * 32 + hi * 8;
                    ai1.q[0] = *(const uint4*)aq1;
                    ai1.q[1] = *(const uint4*)(aq1 + 16);

                    acc0 = __builtin_amdgcn_wmma_f32_16x16x32_f16(false, ai0.v, false, bih.v,
                                                                  (short)0, acc0, false, false);
                    acc1 = __builtin_amdgcn_wmma_f32_16x16x32_f16(false, ai1.v, false, bih.v,
                                                                  (short)0, acc1, false, false);
                }
            }

            // tanh (v_tanh_f32) + pack to f16 (held in regs until barrier)
            v2h hn0[4], hn1[4];
            #pragma unroll
            for (int r = 0; r < 4; ++r) {
                hn0[r] = pack_f16(fast_tanh(acc0[2 * r]), fast_tanh(acc0[2 * r + 1]));
                hn1[r] = pack_f16(fast_tanh(acc1[2 * r]), fast_tanh(acc1[2 * r + 1]));
            }

            // prefetch next layer's recurrent weights into cache hierarchy
            if (l + 1 < LL)
                __builtin_prefetch(whh_g + (size_t)(l + 1) * HH * HH + tid * 64, 0, 1);

            __syncthreads();   // all reads of old h_state[l] complete

            // ---- in-place state update h_state[l] <- tanh(acc) ----
            #pragma unroll
            for (int r = 0; r < 4; ++r) {
                hcur[(size_t)(hi * 8 + 2 * r) * HH + n]          = hn0[r][0];
                hcur[(size_t)(hi * 8 + 2 * r + 1) * HH + n]      = hn0[r][1];
                hcur[(size_t)(16 + hi * 8 + 2 * r) * HH + n]     = hn1[r][0];
                hcur[(size_t)(16 + hi * 8 + 2 * r + 1) * HH + n] = hn1[r][1];
            }
        }
    }
    __syncthreads();

    // ---- head: 1x1 conv over [last_action ; h_last], write raw logits ----
    if (tid < BW) {
        int bg = bs * BW + tid;
        const _Float16* hf = hS + (size_t)(LL - 1) * BW * HH + (size_t)tid * HH;
        float acc = conv_b[0] + conv_w[0] * last_action[bg * (PP + 1) + 1 + p];
        #pragma unroll 4
        for (int h = 0; h < HH; ++h) acc += conv_w[1 + h] * (float)hf[h];
        out[bg * (PP + 1) + 1 + p] = acc;
    }
}

__global__ void eiie_softmax_kernel(float* __restrict__ out)
{
    int b = threadIdx.x;
    if (b >= BB) return;
    float y[PP];
    float m = 0.0f;   // implicit cash logit = 0
    #pragma unroll
    for (int j = 0; j < PP; ++j) {
        y[j] = out[b * (PP + 1) + 1 + j];
        m = fmaxf(m, y[j]);
    }
    float e0 = __expf(-m);
    float d = e0;
    float e[PP];
    #pragma unroll
    for (int j = 0; j < PP; ++j) { e[j] = __expf(y[j] - m); d += e[j]; }
    float inv = 1.0f / d;
    out[b * (PP + 1)] = e0 * inv;
    #pragma unroll
    for (int j = 0; j < PP; ++j) out[b * (PP + 1) + 1 + j] = e[j] * inv;
}

extern "C" void kernel_launch(void* const* d_in, const int* in_sizes, int n_in,
                              void* d_out, int out_size, void* d_ws, size_t ws_size,
                              hipStream_t stream)
{
    const float* obs    = (const float*)d_in[0];
    const float* lastA  = (const float*)d_in[1];
    const float* w_ih0  = (const float*)d_in[2];
    const float* w_ih   = (const float*)d_in[3];
    const float* w_hh   = (const float*)d_in[4];
    const float* b_ih   = (const float*)d_in[5];
    const float* b_hh   = (const float*)d_in[6];
    const float* conv_w = (const float*)d_in[7];
    const float* conv_b = (const float*)d_in[8];
    float* out = (float*)d_out;

    dim3 grid(PP * BSPLIT);
    dim3 block(256);
    eiie_rnn_kernel<<<grid, block, LDS_BYTES, stream>>>(
        obs, lastA, w_ih0, w_ih, w_hh, b_ih, b_hh, conv_w, conv_b, out);
    eiie_softmax_kernel<<<dim3(1), dim3(64), 0, stream>>>(out);
}
